// Grid_Block_56169582297331
// MI455X (gfx1250) — compile-verified, matrix operand order checked by reading, not run on hardware
//
#include <hip/hip_runtime.h>
#include <hip/hip_bf16.h>

// MI455X (gfx1250) implementation of the MaxViT-style grid block.
// All GEMMs (QKV, q@k^T, attn@v, both 3x3 convs as implicit GEMM) run on
// v_wmma_f32_16x16x32_bf16. Activations for the convs are kept as bf16 NHWC
// (67MB -> resident in the 192MB L2). Fragment loads are b128-vectorized:
// each lane's 8 fragment dwords are two contiguous 16B groups of the row's
// 16-dword k-block ({0..3}+sub*4 and {8..11}+sub*4).
//
// Workspace layout (requires ~272MB):
//   [0, 384K)            qkv_w packed bf16   [128 c-pairs][768]
//   [384K, 1.5M)         conv1_w packed bf16 [72 kb][256 cout][16 k-pairs]
//   [1.5M, 2.625M)       conv2_w packed bf16 (same layout)
//   [2.625M, +134M)      u  f32 NCHW  (post-LN2, residual source)
//   [.., +67M)           u  bf16 NHWC (conv1 input)
//   [.., +67M)           m1 bf16 NHWC (conv2 input)

typedef __attribute__((ext_vector_type(16))) __bf16 v16bf;
typedef __attribute__((ext_vector_type(8)))  float  v8f;

union FragBF { unsigned u[8]; v16bf v; };

__device__ __forceinline__ unsigned short f2bf(float f) {
    unsigned u = __builtin_bit_cast(unsigned, f);
    u += 0x7FFFu + ((u >> 16) & 1u);           // round-to-nearest-even
    return (unsigned short)(u >> 16);
}
__device__ __forceinline__ float bf2f(unsigned short h) {
    unsigned u = ((unsigned)h) << 16;
    return __builtin_bit_cast(float, u);
}
__device__ __forceinline__ unsigned pk2(float a, float b) {
    return (unsigned)f2bf(a) | ((unsigned)f2bf(b) << 16);
}
// k offset of the low element of the bf16 pair held in fragment dword j
// (16-bit A/B layout: lanes 0-15 get K {0..7,16..23}, lanes 16-31 get +8).
__device__ __forceinline__ int koff(int j, int sub) {
    return ((j & 3) << 1) + ((j >> 2) << 4) + (sub << 3);
}
// Load a fragment whose 16-dword k-block starts (16B-aligned) at row16:
// two b128 loads per lane.
__device__ __forceinline__ void ldfrag(FragBF& f, const unsigned* row16, int sub) {
    const uint4* p = (const uint4*)row16;
    uint4 lo = p[sub], hi = p[sub + 2];
    f.u[0] = lo.x; f.u[1] = lo.y; f.u[2] = lo.z; f.u[3] = lo.w;
    f.u[4] = hi.x; f.u[5] = hi.y; f.u[6] = hi.z; f.u[7] = hi.w;
}
__device__ __forceinline__ v8f wmma_bf(const FragBF& a, const FragBF& b, v8f c) {
    return __builtin_amdgcn_wmma_f32_16x16x32_bf16(false, a.v, false, b.v,
                                                   (short)0, c, false, false);
}

// ---------------------------------------------------------------------------
// Weight prep: f32 -> bf16 packed consecutive-k pairs per dword.
// qkv:  qkvp[c/2][d]            B[k=c][n=d] = qkv_w[d][c]
// conv: wNp[kb][cout][kpair16]  B[k=(dy*3+dx)*256+cin][n=cout] = w[cout][cin][dy][dx]
// ---------------------------------------------------------------------------
__global__ void __launch_bounds__(256) prep_weights(
    const float* __restrict__ qkv_w, const float* __restrict__ c1w,
    const float* __restrict__ c2w,
    unsigned* __restrict__ qkvp, unsigned* __restrict__ w1p,
    unsigned* __restrict__ w2p) {
    int idx = blockIdx.x * 256 + threadIdx.x;
    if (idx < 98304) {                       // 128 c-pairs * 768
        int cp = idx / 768, d = idx - cp * 768;
        qkvp[idx] = pk2(qkv_w[d * 256 + 2 * cp], qkv_w[d * 256 + 2 * cp + 1]);
        return;
    }
    idx -= 98304;
    const float* w = c1w;
    unsigned* dst = w1p;
    if (idx >= 294912) { idx -= 294912; w = c2w; dst = w2p; }
    if (idx >= 294912) return;
    int kk = idx >> 8, n = idx & 255;        // 1152 k-pairs * 256
    int kb = kk >> 4, kp = kk & 15;          // k-block of 16 pairs (32 k)
    int k0 = kk << 1;
    int dyx = k0 >> 8, cin = k0 & 255;
    dst[(kb * 256 + n) * 16 + kp] = pk2(w[(n * 256 + cin) * 9 + dyx],
                                        w[(n * 256 + cin + 1) * 9 + dyx]);
}

// ---------------------------------------------------------------------------
// Attention kernel: one 8x8 grid window per workgroup (2048 blocks, 8 waves).
// Dynamic LDS 160KB:  [0,96K) qkv bf16 [64][768]
//                     [96K,160K) reused: h u32 [64][128] / attn bf16 8x[64][64]
//                                        / o f32 [64][256]
// ---------------------------------------------------------------------------
__global__ void __launch_bounds__(256) attn_kernel(
    const float* __restrict__ x, const float* __restrict__ ln1g,
    const float* __restrict__ ln1b, const unsigned* __restrict__ qkvp,
    const float* __restrict__ rel, const float* __restrict__ ln2g,
    const float* __restrict__ ln2b, float* __restrict__ u_nchw,
    unsigned* __restrict__ u_nhwc) {
    extern __shared__ char smem[];
    unsigned short* qkvs = (unsigned short*)smem;           // 96KB
    unsigned*       h32  = (unsigned*)(smem + 98304);       // 32KB (phase A/B)
    unsigned short* atts = (unsigned short*)(smem + 98304); // 64KB (phase C/D)
    float*          o32  = (float*)(smem + 98304);          // 64KB (phase E)

    const int tid = threadIdx.x, lane = tid & 31;
    const int wv = tid >> 5, sub = lane >> 4, l15 = lane & 15;
    const int blk = blockIdx.x;
    const int b = blk >> 8, wy = (blk >> 4) & 15, wx = blk & 15;

    // ---- Phase A: load window (grid partition: pixel = (i*16+wy, j*16+wx)),
    //      LN1, store h as packed bf16 pairs. 4 threads per token row.
    {
        int row = tid >> 2, q = tid & 3;
        int yy = (row >> 3) * 16 + wy, xx = (row & 7) * 16 + wx;
        const float* xp = x + ((size_t)b * 256 * 128 + yy) * 128 + xx;
        float s = 0.f, ss = 0.f;
        for (int c = q * 64; c < q * 64 + 64; ++c) {
            float v = xp[(size_t)c * 16384];
            s += v; ss += v * v;
        }
        s  += __shfl_xor(s, 1, 4);  s  += __shfl_xor(s, 2, 4);
        ss += __shfl_xor(ss, 1, 4); ss += __shfl_xor(ss, 2, 4);
        float mean = s * (1.f / 256.f);
        float rstd = rsqrtf(ss * (1.f / 256.f) - mean * mean + 1e-5f);
        for (int cp = 0; cp < 32; ++cp) {
            int c = q * 64 + cp * 2;
            float v0 = (xp[(size_t)c * 16384] - mean) * rstd * ln1g[c] + ln1b[c];
            float v1 = (xp[(size_t)(c + 1) * 16384] - mean) * rstd * ln1g[c + 1] + ln1b[c + 1];
            h32[row * 128 + (c >> 1)] = pk2(v0, v1);
        }
    }
    __syncthreads();

    // ---- Phase B: qkv = h @ qkv_w^T  (M=64,N=768,K=256) -> bf16 in LDS.
    {
        int mb = (wv >> 1) * 16;
        int ntg = wv & 1;
        for (int t = 0; t < 24; ++t) {
            int nb = (ntg * 24 + t) * 16;
            v8f acc = {};
            for (int kb = 0; kb < 8; ++kb) {
                FragBF a, bm;
                ldfrag(a, h32 + (mb + l15) * 128 + kb * 16, sub);
#pragma unroll
                for (int j = 0; j < 8; ++j) {
                    int k = kb * 32 + koff(j, sub);
                    bm.u[j] = qkvp[(k >> 1) * 768 + nb + l15];
                }
                acc = wmma_bf(a, bm, acc);
            }
#pragma unroll
            for (int r = 0; r < 8; ++r)
                qkvs[(mb + r + sub * 8) * 768 + nb + l15] = f2bf(acc[r]);
        }
    }
    __syncthreads();

    // ---- Phase B2: cosine-normalize q and k rows (per token, per head).
    for (int t = tid; t < 1024; t += 256) {
        int qk = t >> 9, rh = t & 511, row = rh >> 3, hh = rh & 7;
        int base = row * 768 + qk * 256 + hh * 32;
        float ss = 0.f;
        for (int e = 0; e < 32; ++e) { float v = bf2f(qkvs[base + e]); ss += v * v; }
        float rn = rsqrtf(ss);
        for (int e = 0; e < 32; ++e)
            qkvs[base + e] = f2bf(bf2f(qkvs[base + e]) * rn);
    }
    __syncthreads();

    // ---- Phase C: one head per wave. logits = scale*(q@k^T) + relbias,
    //      row softmax via shfl-xor(width 16) matching D-tile layout.
    {
        const int h = wv;
        const unsigned* q32 = (const unsigned*)qkvs;
        v8f acc[4][4];
        for (int mt = 0; mt < 4; ++mt) {
            FragBF a;
            ldfrag(a, q32 + (mt * 16 + l15) * 384 + h * 16, sub);
            for (int nt = 0; nt < 4; ++nt) {
                FragBF bm;
                ldfrag(bm, q32 + (nt * 16 + l15) * 384 + 128 + h * 16, sub);
                v8f z = {};
                acc[mt][nt] = wmma_bf(a, bm, z);
            }
        }
        const float scale = 0.17677669529663687f;  // 1/sqrt(32)
        for (int mt = 0; mt < 4; ++mt) {
#pragma unroll
            for (int r = 0; r < 8; ++r) {
                int n = mt * 16 + r + sub * 8;
                int i1 = n >> 3, j1 = n & 7;
                float vals[4];
                float mx = -3.4e38f;
                for (int nt = 0; nt < 4; ++nt) {
                    int m = nt * 16 + l15;
                    int idx = (i1 - (m >> 3) + 7) * 15 + (j1 - (m & 7) + 7);
                    float v = acc[mt][nt][r] * scale + rel[idx * 8 + h];
                    vals[nt] = v;
                    mx = fmaxf(mx, v);
                }
                mx = fmaxf(mx, __shfl_xor(mx, 1, 16));
                mx = fmaxf(mx, __shfl_xor(mx, 2, 16));
                mx = fmaxf(mx, __shfl_xor(mx, 4, 16));
                mx = fmaxf(mx, __shfl_xor(mx, 8, 16));
                float sum = 0.f;
                for (int nt = 0; nt < 4; ++nt) { vals[nt] = expf(vals[nt] - mx); sum += vals[nt]; }
                sum += __shfl_xor(sum, 1, 16);
                sum += __shfl_xor(sum, 2, 16);
                sum += __shfl_xor(sum, 4, 16);
                sum += __shfl_xor(sum, 8, 16);
                float inv = 1.f / sum;
                for (int nt = 0; nt < 4; ++nt)
                    atts[wv * 4096 + n * 64 + nt * 16 + l15] = f2bf(vals[nt] * inv);
            }
        }
    }
    __syncthreads();

    // ---- Phase D: o = attn @ v  (M=64,N=32,K=64) per head.
    {
        const int h = wv;
        const unsigned* a32 = (const unsigned*)(atts + wv * 4096);
        v8f zero = {};
        v8f acc2[4][2];
        for (int mt = 0; mt < 4; ++mt)
            for (int nt = 0; nt < 2; ++nt) acc2[mt][nt] = zero;
        for (int mt = 0; mt < 4; ++mt) {
            for (int kb = 0; kb < 2; ++kb) {
                FragBF a;
                ldfrag(a, a32 + (mt * 16 + l15) * 32 + kb * 16, sub);
                for (int nt = 0; nt < 2; ++nt) {
                    FragBF bm;
#pragma unroll
                    for (int j = 0; j < 8; ++j) {
                        int k = kb * 32 + koff(j, sub);
                        int col = 512 + h * 32 + nt * 16 + l15;
                        unsigned lo = qkvs[k * 768 + col];
                        unsigned hi = qkvs[(k + 1) * 768 + col];
                        bm.u[j] = lo | (hi << 16);
                    }
                    acc2[mt][nt] = wmma_bf(a, bm, acc2[mt][nt]);
                }
            }
        }
        __syncthreads();   // everyone done reading attn before o32 overwrites it
        for (int mt = 0; mt < 4; ++mt) {
            for (int nt = 0; nt < 2; ++nt) {
#pragma unroll
                for (int r = 0; r < 8; ++r)
                    o32[(mt * 16 + r + sub * 8) * 256 + h * 32 + nt * 16 + l15] =
                        acc2[mt][nt][r];
            }
        }
    }
    __syncthreads();

    // ---- Phase E: out = g + o, LN2, write u (f32 NCHW + bf16 NHWC).
    {
        int row = tid >> 2, q = tid & 3;
        int yy = (row >> 3) * 16 + wy, xx = (row & 7) * 16 + wx;
        const float* xp = x + ((size_t)b * 256 * 128 + yy) * 128 + xx;
        float s = 0.f, ss = 0.f;
        for (int c = q * 64; c < q * 64 + 64; ++c) {
            float v = xp[(size_t)c * 16384] + o32[row * 256 + c];
            o32[row * 256 + c] = v;
            s += v; ss += v * v;
        }
        s  += __shfl_xor(s, 1, 4);  s  += __shfl_xor(s, 2, 4);
        ss += __shfl_xor(ss, 1, 4); ss += __shfl_xor(ss, 2, 4);
        float mean = s * (1.f / 256.f);
        float rstd = rsqrtf(ss * (1.f / 256.f) - mean * mean + 1e-5f);
        size_t base0 = (size_t)b * 4194304 + (size_t)yy * 128 + xx;  // NCHW, c=0
        size_t pix = ((size_t)b * 128 + yy) * 128 + xx;
        for (int cp = 0; cp < 32; ++cp) {
            int c = q * 64 + cp * 2;
            float v0 = (o32[row * 256 + c] - mean) * rstd * ln2g[c] + ln2b[c];
            float v1 = (o32[row * 256 + c + 1] - mean) * rstd * ln2g[c + 1] + ln2b[c + 1];
            u_nchw[base0 + (size_t)c * 16384] = v0;
            u_nchw[base0 + (size_t)(c + 1) * 16384] = v1;
            u_nhwc[pix * 128 + (c >> 1)] = pk2(v0, v1);
        }
    }
}

// ---------------------------------------------------------------------------
// Implicit-GEMM 3x3 conv + BN + GELU (+ optional residual & f32 NCHW output).
// One workgroup per 64-pixel half-row x 256 channels; K = 9*256 = 2304.
// A: two b128 loads/lane from the L2-resident bf16 NHWC tensor.
// B: weight k-block (16KB) double-buffered through LDS, two ds_load_b128
//    per fragment. One __syncthreads per k-step.
// ---------------------------------------------------------------------------
__global__ void __launch_bounds__(256) conv_kernel(
    const unsigned* __restrict__ src_nhwc, const uint4* __restrict__ wpack4,
    const float* __restrict__ bg, const float* __restrict__ bb,
    const float* __restrict__ bm, const float* __restrict__ bv,
    const float* __restrict__ resid, float* __restrict__ out_f32,
    unsigned short* __restrict__ out_bf16) {
    __shared__ uint4 wt[2][1024];          // 2 x 16KB weight k-block

    const int blk = blockIdx.x;
    const int x0 = (blk & 1) * 64, y = (blk >> 1) & 127, b = blk >> 8;
    const int tid = threadIdx.x, lane = tid & 31;
    const int wv = tid >> 5, sub = lane >> 4, l15 = lane & 15;
    const int mt = wv >> 1, ntb = (wv & 1) * 128;
    const int xr = x0 + mt * 16 + l15;     // pixel x of this lane's A-row

    // preload k-block 0
#pragma unroll
    for (int i = 0; i < 4; ++i) wt[0][tid + i * 256] = wpack4[tid + i * 256];
    __syncthreads();

    v8f zero = {};
    v8f acc[8];
    for (int i = 0; i < 8; ++i) acc[i] = zero;

    for (int kb = 0; kb < 72; ++kb) {
        int cur = kb & 1;
        if (kb < 71) {                     // stage next weight k-block
            const uint4* s = wpack4 + (size_t)(kb + 1) * 1024;
#pragma unroll
            for (int i = 0; i < 4; ++i) wt[cur ^ 1][tid + i * 256] = s[tid + i * 256];
        }
        int dyx = kb >> 3;
        int dy = dyx / 3 - 1, dx = dyx - (dyx / 3) * 3 - 1;
        int c0h = ((kb & 7) * 32) >> 1;    // u32 channel-pair base
        int ys = y + dy, xs = xr + dx;
        bool ok = ((unsigned)ys < 128u) && ((unsigned)xs < 128u);
        FragBF a;
        {
            const uint4* ap4 = (const uint4*)(src_nhwc +
                (((size_t)b * 128 + ys) * 128 + xs) * 128 + c0h);
            uint4 lo, hi;
            if (ok) { lo = ap4[sub]; hi = ap4[sub + 2]; }
            else    { lo = make_uint4(0u, 0u, 0u, 0u); hi = lo; }
            a.u[0] = lo.x; a.u[1] = lo.y; a.u[2] = lo.z; a.u[3] = lo.w;
            a.u[4] = hi.x; a.u[5] = hi.y; a.u[6] = hi.z; a.u[7] = hi.w;
        }
        if (kb < 71) {                     // prefetch next A line into cache
            int dyx2 = (kb + 1) >> 3;
            int dy2 = dyx2 / 3 - 1, dx2 = dyx2 - (dyx2 / 3) * 3 - 1;
            int c2h = (((kb + 1) & 7) * 32) >> 1;
            int ys2 = y + dy2, xs2 = xr + dx2;
            if (((unsigned)ys2 < 128u) && ((unsigned)xs2 < 128u))
                __builtin_prefetch(src_nhwc +
                    (((size_t)b * 128 + ys2) * 128 + xs2) * 128 + c2h, 0, 3);
        }
        const unsigned* wl = (const unsigned*)wt[cur];
        for (int nt = 0; nt < 8; ++nt) {
            FragBF w;
            ldfrag(w, wl + (ntb + nt * 16 + l15) * 16, sub);
            acc[nt] = wmma_bf(a, w, acc[nt]);
        }
        __syncthreads();                   // reads of cur done, nxt staged
    }

    for (int nt = 0; nt < 8; ++nt) {
        int ch = ntb + nt * 16 + l15;
        float g = bg[ch], be = bb[ch], mn = bm[ch];
        float rs = rsqrtf(bv[ch] + 1e-5f);
#pragma unroll
        for (int r = 0; r < 8; ++r) {
            int xp = x0 + mt * 16 + r + sub * 8;
            float v = (acc[nt][r] - mn) * rs * g + be;
            v = 0.5f * v * (1.f + erff(v * 0.70710678118654752f));  // exact GELU
            size_t pix = ((size_t)b * 128 + y) * 128 + xp;
            if (out_bf16) out_bf16[pix * 256 + ch] = f2bf(v);
            if (out_f32) {
                size_t adr = (((size_t)b * 256 + ch) * 128 + y) * 128 + xp;
                out_f32[adr] = v + resid[adr];
            }
        }
    }
}

// ---------------------------------------------------------------------------
extern "C" void kernel_launch(void* const* d_in, const int* in_sizes, int n_in,
                              void* d_out, int out_size, void* d_ws,
                              size_t ws_size, hipStream_t stream) {
    const float* x     = (const float*)d_in[0];
    const float* ln1g  = (const float*)d_in[1];
    const float* ln1b  = (const float*)d_in[2];
    const float* qkv_w = (const float*)d_in[3];
    const float* rel   = (const float*)d_in[4];
    const float* ln2g  = (const float*)d_in[5];
    const float* ln2b  = (const float*)d_in[6];
    const float* c1w   = (const float*)d_in[7];
    const float* bn1g  = (const float*)d_in[8];
    const float* bn1b  = (const float*)d_in[9];
    const float* bn1m  = (const float*)d_in[10];
    const float* bn1v  = (const float*)d_in[11];
    const float* c2w   = (const float*)d_in[12];
    const float* bn2g  = (const float*)d_in[13];
    const float* bn2b  = (const float*)d_in[14];
    const float* bn2m  = (const float*)d_in[15];
    const float* bn2v  = (const float*)d_in[16];

    char* ws = (char*)d_ws;
    unsigned*       qkvp   = (unsigned*)(ws);
    uint4*          w1p4   = (uint4*)(ws + 393216);
    uint4*          w2p4   = (uint4*)(ws + 1572864);
    float*          u_nchw = (float*)(ws + 2752512);
    unsigned*       u_nhwc = (unsigned*)(ws + 136970240);
    unsigned short* m1     = (unsigned short*)(ws + 204079104);

    prep_weights<<<2688, 256, 0, stream>>>(qkv_w, c1w, c2w, qkvp,
                                           (unsigned*)w1p4, (unsigned*)w2p4);
    attn_kernel<<<2048, 256, 163840, stream>>>(x, ln1g, ln1b, qkvp, rel,
                                               ln2g, ln2b, u_nchw, u_nhwc);
    conv_kernel<<<2048, 256, 0, stream>>>(u_nhwc, w1p4, bn1g, bn1b, bn1m, bn1v,
                                          nullptr, nullptr, m1);
    conv_kernel<<<2048, 256, 0, stream>>>((const unsigned*)m1, w2p4, bn2g, bn2b,
                                          bn2m, bn2v, u_nchw, (float*)d_out,
                                          nullptr);
}
